// SANNetwork_13924283973994
// MI455X (gfx1250) — matrix operand order repaired and checked by reading, not compiled, not run on hardware
//
#include <hip/hip_runtime.h>
#include <hip/hip_bf16.h>
#include <math.h>

// Problem dims (fixed by the reference)
#define BB   128
#define TT   256
#define EE   512
#define HH   512
#define OUTC 2

typedef __attribute__((ext_vector_type(16))) __bf16 v16bf;
typedef __attribute__((ext_vector_type(8)))  float  v8f;

__device__ __forceinline__ unsigned short f2bf(float f) {
    union { float f; unsigned u; } x; x.f = f;
    unsigned r = x.u + 0x7FFFu + ((x.u >> 16) & 1u);   // round-to-nearest-even
    return (unsigned short)(r >> 16);
}
__device__ __forceinline__ float bf2f(unsigned short b) {
    union { unsigned u; float f; } x; x.u = ((unsigned)b) << 16;
    return x.f;
}

// 32-byte fragment: 16 bf16 values for one WMMA operand, per lane.
union FragBF {
    uint4 q[2];
    v16bf v;
};

#define WMMA_BF16(A, B, C) \
    __builtin_amdgcn_wmma_f32_16x16x32_bf16(false, (A), false, (B), (short)0, (C), false, false)

// ---------------------------------------------------------------- utility
__global__ void cvt_f32_bf16(const float* __restrict__ in,
                             unsigned short* __restrict__ out, int n) {
    int i = blockIdx.x * blockDim.x + threadIdx.x;
    if (i < n) out[i] = f2bf(in[i]);
}

__global__ void zero_f32(float* __restrict__ p, int n) {
    int i = blockIdx.x * blockDim.x + threadIdx.x;
    if (i < n) p[i] = 0.f;
}
__global__ void zero_b16(unsigned short* __restrict__ p, int n) {
    int i = blockIdx.x * blockDim.x + threadIdx.x;
    if (i < n) p[i] = 0;
}

// Gather embedding rows -> bf16 [B,T,E] flat
__global__ void gather_embed(const int* __restrict__ tok,
                             const float* __restrict__ embed,
                             unsigned short* __restrict__ out, int n) {
    int i = blockIdx.x * blockDim.x + threadIdx.x;
    if (i >= n) return;
    int d  = i & (EE - 1);
    int bt = i >> 9;            // E = 512
    int t  = tok[bt];
    out[i] = f2bf(embed[(size_t)t * EE + d]);
}

// ---------------------------------------------------------------- GEMM
// out[M,N] = A[M,K](bf16,row-major) x W[N,K](bf16, torch layout)^T + bias
// Register-blocked: one wave computes a 32x64 macro-tile (2x4 WMMA tiles).
// Per K-step: 2 A frags + 4 B frags = 12 b128 loads feeding 8 WMMAs.
__global__ void gemm_bf16_wmma(const unsigned short* __restrict__ A,
                               const unsigned short* __restrict__ W,
                               const float* __restrict__ bias,
                               unsigned short* __restrict__ outb,
                               float* __restrict__ outf,
                               int M, int N, int K) {
    int lane = threadIdx.x & 31;
    int wid  = blockIdx.x * (blockDim.x >> 5) + (threadIdx.x >> 5);
    int tilesN = N >> 6;                 // 64-wide macro columns
    int tm = wid / tilesN;               // 32-tall macro row
    int tn = wid - tm * tilesN;
    if (tm * 32 >= M) return;

    int ka = (lane >> 4) * 16;
    int kb = (lane >> 4) * 16;
    const unsigned short* Ap0 = A + (size_t)(tm * 32 + (lane & 15)) * K;
    const unsigned short* Ap1 = Ap0 + (size_t)16 * K;
    const unsigned short* Wp  = W + (size_t)(tn * 64 + (lane & 15)) * K;

    v8f acc[2][4] = {};
    for (int k = 0; k < K; k += 32) {
        FragBF a0, a1, b[4];
        const uint4* p;
        p = (const uint4*)(Ap0 + k + ka); a0.q[0] = p[0]; a0.q[1] = p[1];
        p = (const uint4*)(Ap1 + k + ka); a1.q[0] = p[0]; a1.q[1] = p[1];
#pragma unroll
        for (int j = 0; j < 4; ++j) {
            p = (const uint4*)(Wp + (size_t)j * 16 * K + k + kb);
            b[j].q[0] = p[0]; b[j].q[1] = p[1];
        }
#pragma unroll
        for (int j = 0; j < 4; ++j) {
            acc[0][j] = WMMA_BF16(a0.v, b[j].v, acc[0][j]);
            acc[1][j] = WMMA_BF16(a1.v, b[j].v, acc[1][j]);
        }
    }

    int ro = (lane >> 4) * 8;
#pragma unroll
    for (int i = 0; i < 2; ++i) {
#pragma unroll
        for (int j = 0; j < 4; ++j) {
            int col = tn * 64 + j * 16 + (lane & 15);
            float bv = bias ? bias[col] : 0.f;
#pragma unroll
            for (int v = 0; v < 8; ++v) {
                int row = tm * 32 + i * 16 + ro + v;
                size_t idx = (size_t)row * N + col;
                float val = acc[i][j][v] + bv;
                if (outf) outf[idx] = val;
                else      outb[idx] = f2bf(val);
            }
        }
    }
}

// ---------------------------------------------------------------- fused GRU step
// One wave: 32-row group (2 row tiles) x one 16-col triple {n, H+n, 2H+n}.
// Per K-step: 2 A frags + 3 B frags = 10 b128 loads feeding 6 WMMAs; the three
// gate tiles live in one wave so gating fuses without any grid sync.
__global__ void gru_step_kernel(const unsigned short* __restrict__ hb,   // [B,H] bf16 (h_t)
                                const unsigned short* __restrict__ Whb,  // [3H,H] bf16
                                const float* __restrict__ bh,            // [3H]
                                const unsigned short* __restrict__ xw,   // [B*T,3H] bf16 (x@Wi^T+bi)
                                const float* __restrict__ hf,            // [B,H] f32 (h_t)
                                float* __restrict__ hf_out,              // [B,H] f32 (h_{t+1})
                                unsigned short* __restrict__ hb_out,     // [B,H] bf16
                                unsigned short* __restrict__ hs_out,     // [B,T,H] or null
                                int t) {
    const int H = HH, T = TT;
    int lane = threadIdx.x & 31;
    int wid  = blockIdx.x * (blockDim.x >> 5) + (threadIdx.x >> 5);
    int rt = wid >> 5;       // 0..3  (32-row groups of B=128)
    int ct = wid & 31;       // 0..31 (16-col groups of H=512)

    int ka = (lane >> 4) * 16;
    int kb = (lane >> 4) * 16;
    int n  = ct * 16 + (lane & 15);
    const unsigned short* Ap0 = hb + (size_t)(rt * 32 + (lane & 15)) * H;
    const unsigned short* Ap1 = Ap0 + (size_t)16 * H;
    const unsigned short* W0 = Whb + (size_t)(0 * H + n) * H;
    const unsigned short* W1 = Whb + (size_t)(1 * H + n) * H;
    const unsigned short* W2 = Whb + (size_t)(2 * H + n) * H;

    v8f acc[2][3] = {};
    for (int k = 0; k < H; k += 32) {
        FragBF a0, a1, b0, b1, b2;
        const uint4* p;
        p = (const uint4*)(Ap0 + k + ka); a0.q[0] = p[0]; a0.q[1] = p[1];
        p = (const uint4*)(Ap1 + k + ka); a1.q[0] = p[0]; a1.q[1] = p[1];
        p = (const uint4*)(W0 + k + kb);  b0.q[0] = p[0]; b0.q[1] = p[1];
        p = (const uint4*)(W1 + k + kb);  b1.q[0] = p[0]; b1.q[1] = p[1];
        p = (const uint4*)(W2 + k + kb);  b2.q[0] = p[0]; b2.q[1] = p[1];
        acc[0][0] = WMMA_BF16(a0.v, b0.v, acc[0][0]);
        acc[1][0] = WMMA_BF16(a1.v, b0.v, acc[1][0]);
        acc[0][1] = WMMA_BF16(a0.v, b1.v, acc[0][1]);
        acc[1][1] = WMMA_BF16(a1.v, b1.v, acc[1][1]);
        acc[0][2] = WMMA_BF16(a0.v, b2.v, acc[0][2]);
        acc[1][2] = WMMA_BF16(a1.v, b2.v, acc[1][2]);
    }

    float bhr = bh[n], bhz = bh[H + n], bhn = bh[2 * H + n];
    int ro = (lane >> 4) * 8;
#pragma unroll
    for (int i = 0; i < 2; ++i) {
#pragma unroll
        for (int v = 0; v < 8; ++v) {
            int b = rt * 32 + i * 16 + ro + v;
            size_t xrow = ((size_t)b * T + t) * (3 * H);
            float xr = bf2f(xw[xrow + n]);
            float xz = bf2f(xw[xrow + H + n]);
            float xn = bf2f(xw[xrow + 2 * H + n]);
            float r  = 1.f / (1.f + __expf(-(xr + acc[i][0][v] + bhr)));
            float z  = 1.f / (1.f + __expf(-(xz + acc[i][1][v] + bhz)));
            float nn = tanhf(xn + r * (acc[i][2][v] + bhn));
            float hold = hf[(size_t)b * H + n];
            float hnew = (1.f - z) * nn + z * hold;
            hf_out[(size_t)b * H + n] = hnew;
            hb_out[(size_t)b * H + n] = f2bf(hnew);
            if (hs_out) hs_out[((size_t)b * T + t) * H + n] = f2bf(hnew);
        }
    }
}

// ---------------------------------------------------------------- attention pieces
__global__ void sum_over_t(const unsigned short* __restrict__ u,  // [B,T,D] bf16
                           float* __restrict__ usum, int D) {     // [B,D]
    int i = blockIdx.x * blockDim.x + threadIdx.x;
    if (i >= BB * D) return;
    int b = i / D, d = i - b * D;
    float s = 0.f;
    const unsigned short* p = u + (size_t)b * TT * D + d;
    for (int t = 0; t < TT; ++t) s += bf2f(p[(size_t)t * D]);
    usum[i] = s;
}

// One wave per (b,t): w = ue . r; sc = softmax_D(tanh(w)*Wsc); ctx = sc*usum*r
__global__ void dual_attention_kernel(const float* __restrict__ ue,          // [B*T,D]
                                      const unsigned short* __restrict__ r,  // [B*T,D] bf16
                                      const float* __restrict__ usum,        // [B,D]
                                      const float* __restrict__ Wsc,         // [D]
                                      unsigned short* __restrict__ vout,     // [B*T,vstride]
                                      int D, int vstride, int voff) {
    int lane = threadIdx.x & 31;
    int bt = blockIdx.x * (blockDim.x >> 5) + (threadIdx.x >> 5);
    if (bt >= BB * TT) return;
    int b = bt / TT;
    const float* uep = ue + (size_t)bt * D;
    const unsigned short* rp = r + (size_t)bt * D;

    float w = 0.f;
    for (int d = lane; d < D; d += 32) w += uep[d] * bf2f(rp[d]);
#pragma unroll
    for (int o = 16; o > 0; o >>= 1) w += __shfl_xor(w, o, 32);
    float s = tanhf(w);

    float mx = -1e30f;
    for (int d = lane; d < D; d += 32) mx = fmaxf(mx, s * Wsc[d]);
#pragma unroll
    for (int o = 16; o > 0; o >>= 1) mx = fmaxf(mx, __shfl_xor(mx, o, 32));
    float se = 0.f;
    for (int d = lane; d < D; d += 32) se += __expf(s * Wsc[d] - mx);
#pragma unroll
    for (int o = 16; o > 0; o >>= 1) se += __shfl_xor(se, o, 32);
    float inv = 1.f / se;

    const float* up = usum + (size_t)b * D;
    unsigned short* vp = vout + (size_t)bt * vstride + voff;
    for (int d = lane; d < D; d += 32) {
        float sc = __expf(s * Wsc[d] - mx) * inv;
        vp[d] = f2bf(sc * up[d] * bf2f(rp[d]));
    }
}

// ---------------------------------------------------------------- classifier
__global__ void final_softmax(const float* __restrict__ h,    // [B,H] f32
                              const float* __restrict__ Wf,   // [2,H]
                              const float* __restrict__ bfv,  // [2]
                              float* __restrict__ out) {      // [B,2]
    int lane = threadIdx.x & 31;
    int b = blockIdx.x * (blockDim.x >> 5) + (threadIdx.x >> 5);
    if (b >= BB) return;
    const float* hp = h + (size_t)b * HH;
    float a0 = 0.f, a1 = 0.f;
    for (int d = lane; d < HH; d += 32) {
        float hv = hp[d];
        a0 += hv * Wf[d];
        a1 += hv * Wf[HH + d];
    }
#pragma unroll
    for (int o = 16; o > 0; o >>= 1) {
        a0 += __shfl_xor(a0, o, 32);
        a1 += __shfl_xor(a1, o, 32);
    }
    if (lane == 0) {
        float l0 = a0 + bfv[0], l1 = a1 + bfv[1];
        float m  = fmaxf(l0, l1);
        float e0 = __expf(l0 - m), e1 = __expf(l1 - m);
        float sn = e0 + e1;
        out[(size_t)b * 2]     = e0 / sn;
        out[(size_t)b * 2 + 1] = e1 / sn;
    }
}

// ---------------------------------------------------------------- host driver
extern "C" void kernel_launch(void* const* d_in, const int* in_sizes, int n_in,
                              void* d_out, int out_size, void* d_ws, size_t ws_size,
                              hipStream_t stream) {
    (void)in_sizes; (void)n_in; (void)out_size; (void)ws_size;
    const int*   ask   = (const int*)d_in[0];
    const int*   ans   = (const int*)d_in[1];
    const int*   askkw = (const int*)d_in[2];
    const int*   anskw = (const int*)d_in[3];
    const float* embed = (const float*)d_in[4];
    const float* Wi1   = (const float*)d_in[5];
    const float* Wh1   = (const float*)d_in[6];
    const float* bi1   = (const float*)d_in[7];
    const float* bh1   = (const float*)d_in[8];
    const float* Aw    = (const float*)d_in[9];
    const float* bw    = (const float*)d_in[10];
    const float* Ww    = (const float*)d_in[11];
    const float* As    = (const float*)d_in[12];
    const float* bs    = (const float*)d_in[13];
    const float* Ws    = (const float*)d_in[14];
    const float* Wi2   = (const float*)d_in[15];
    const float* Wh2   = (const float*)d_in[16];
    const float* bi2   = (const float*)d_in[17];
    const float* bh2   = (const float*)d_in[18];
    const float* Wf    = (const float*)d_in[19];
    const float* bfv   = (const float*)d_in[20];
    float* out = (float*)d_out;

    const size_t BT  = (size_t)BB * TT;       // 32768
    const size_t BTE = BT * EE;               // 16.78M
    const int    M   = (int)BT;

    char* ws = (char*)d_ws;
    auto carve = [&](size_t bytes) -> char* {
        char* p = ws;
        ws += (bytes + 255) & ~(size_t)255;
        return p;
    };
    unsigned short* ask_e  = (unsigned short*)carve(BTE * 2);
    unsigned short* ans_e  = (unsigned short*)carve(BTE * 2);
    unsigned short* akw    = (unsigned short*)carve(BTE * 2);
    unsigned short* rkw    = (unsigned short*)carve(BTE * 2);
    unsigned short* h_ask  = (unsigned short*)carve(BT * HH * 2);
    unsigned short* h_ans  = (unsigned short*)carve(BT * HH * 2);
    unsigned short* xw_b   = (unsigned short*)carve(BT * 3 * HH * 2);   // reused 3x
    float*          ue_f   = (float*)carve(BT * EE * 4);                // reused 2x
    unsigned short* vbuf   = (unsigned short*)carve(BT * (EE + HH) * 2);
    float*          usum   = (float*)carve((size_t)BB * EE * 4);
    float*          hping_f = (float*)carve((size_t)BB * HH * 4);
    float*          hpong_f = (float*)carve((size_t)BB * HH * 4);
    unsigned short* hping_b = (unsigned short*)carve((size_t)BB * HH * 2);
    unsigned short* hpong_b = (unsigned short*)carve((size_t)BB * HH * 2);
    unsigned short* Wi1b = (unsigned short*)carve((size_t)3 * HH * EE * 2);
    unsigned short* Wh1b = (unsigned short*)carve((size_t)3 * HH * HH * 2);
    unsigned short* Awb  = (unsigned short*)carve((size_t)EE * EE * 2);
    unsigned short* Asb  = (unsigned short*)carve((size_t)HH * HH * 2);
    unsigned short* Wi2b = (unsigned short*)carve((size_t)3 * HH * (EE + HH) * 2);
    unsigned short* Wh2b = (unsigned short*)carve((size_t)3 * HH * HH * 2);

    auto cvt = [&](const float* src, unsigned short* dst, int n) {
        cvt_f32_bf16<<<(n + 255) / 256, 256, 0, stream>>>(src, dst, n);
    };
    auto gemm = [&](const unsigned short* A, const unsigned short* Wb, const float* bias,
                    unsigned short* ob, float* of, int m, int n, int k) {
        int tiles = (m / 32) * (n / 64);       // 32x64 macro-tiles, one per wave
        gemm_bf16_wmma<<<(tiles + 7) / 8, 256, 0, stream>>>(A, Wb, bias, ob, of, m, n, k);
    };
    auto run_gru = [&](const unsigned short* xwbuf, const unsigned short* Whb,
                       const float* bh_, unsigned short* hs) {
        int nh = BB * HH;
        zero_f32<<<(nh + 255) / 256, 256, 0, stream>>>(hping_f, nh);
        zero_b16<<<(nh + 255) / 256, 256, 0, stream>>>(hping_b, nh);
        for (int t = 0; t < TT; ++t) {
            const float*          hin_f  = (t & 1) ? hpong_f : hping_f;
            float*                hout_f = (t & 1) ? hping_f : hpong_f;
            const unsigned short* hin_b  = (t & 1) ? hpong_b : hping_b;
            unsigned short*       hout_b = (t & 1) ? hping_b : hpong_b;
            // waves = (B/32=4) row groups x (H/16=32) col triples = 128 -> 16 blocks
            gru_step_kernel<<<16, 256, 0, stream>>>(hin_b, Whb, bh_, xwbuf,
                                                    hin_f, hout_f, hout_b, hs, t);
        }
        // T even -> final hidden state lands in hping_f
    };

    // 1. weights -> bf16 (original [N,K] layout doubles as WMMA B operand)
    cvt(Wi1, Wi1b, 3 * HH * EE);
    cvt(Wh1, Wh1b, 3 * HH * HH);
    cvt(Aw,  Awb,  EE * EE);
    cvt(As,  Asb,  HH * HH);
    cvt(Wi2, Wi2b, 3 * HH * (EE + HH));
    cvt(Wh2, Wh2b, 3 * HH * HH);

    // 2. embedding gathers -> bf16
    gather_embed<<<(int)(BTE / 256), 256, 0, stream>>>(ask,   embed, ask_e, (int)BTE);
    gather_embed<<<(int)(BTE / 256), 256, 0, stream>>>(ans,   embed, ans_e, (int)BTE);
    gather_embed<<<(int)(BTE / 256), 256, 0, stream>>>(askkw, embed, akw,   (int)BTE);
    gather_embed<<<(int)(BTE / 256), 256, 0, stream>>>(anskw, embed, rkw,   (int)BTE);

    // 3. GRU1 over ask
    gemm(ask_e, Wi1b, bi1, xw_b, nullptr, M, 3 * HH, EE);
    run_gru(xw_b, Wh1b, bh1, h_ask);

    // 4. GRU1 over answer
    gemm(ans_e, Wi1b, bi1, xw_b, nullptr, M, 3 * HH, EE);
    run_gru(xw_b, Wh1b, bh1, h_ans);

    // 5. word-level dual attention -> v[:, 0:512]
    gemm(akw, Awb, bw, nullptr, ue_f, M, EE, EE);
    sum_over_t<<<(BB * EE) / 256, 256, 0, stream>>>(akw, usum, EE);
    dual_attention_kernel<<<(int)(BT / 8), 256, 0, stream>>>(ue_f, rkw, usum, Ww,
                                                             vbuf, EE, EE + HH, 0);

    // 6. sequence-level dual attention -> v[:, 512:1024]
    gemm(h_ask, Asb, bs, nullptr, ue_f, M, HH, HH);
    sum_over_t<<<(BB * HH) / 256, 256, 0, stream>>>(h_ask, usum, HH);
    dual_attention_kernel<<<(int)(BT / 8), 256, 0, stream>>>(ue_f, h_ans, usum, Ws,
                                                             vbuf, HH, EE + HH, EE);

    // 7. GRU2 over v (keep only final hidden state)
    gemm(vbuf, Wi2b, bi2, xw_b, nullptr, M, 3 * HH, EE + HH);
    run_gru(xw_b, Wh2b, bh2, nullptr);

    // 8. classifier + softmax
    final_softmax<<<(BB / 8), 256, 0, stream>>>(hping_f, Wf, bfv, out);
}
